// ImprovedSSM_61641370632867
// MI455X (gfx1250) — matrix operand order repaired and checked by reading, not compiled, not run on hardware
//
#include <hip/hip_runtime.h>
#include <stdint.h>

typedef __attribute__((ext_vector_type(16))) _Float16 v16h;
typedef __attribute__((ext_vector_type(8)))  float    v8f;

#define B_    2
#define S_    2048
#define H_    1024
#define I_    2048
#define NST   16
#define ROWS  (B_ * S_)      // 4096
#define TWO_I (2 * I_)       // 4096

// ---------------------------------------------------------------------------
// CDNA5 async copy helpers (ISA 08_async_tensor: GLOBAL_LOAD_ASYNC_TO_LDS_*)
// vdst = VGPR holding LDS byte offset, vaddr = 64-bit global address (GV mode).
// Tracked by ASYNCcnt; async loads retire in order -> s_wait_asynccnt N works
// as a sliding window for double buffering.
// ---------------------------------------------------------------------------
__device__ __forceinline__ void async_cp_b128(unsigned lds_off, const void* g) {
    asm volatile("global_load_async_to_lds_b128 %0, %1, off"
                 :: "v"(lds_off), "v"(g) : "memory");
}
__device__ __forceinline__ void wait_async_le4() {
    asm volatile("s_wait_asynccnt 0x4" ::: "memory");
}
__device__ __forceinline__ void wait_async_0() {
    asm volatile("s_wait_asynccnt 0x0" ::: "memory");
}

// ---------------------------------------------------------------------------
// Kernel 0: A2[i,n] = -clip(softplus(A_param[i,n]), 0.1, 10)
// ---------------------------------------------------------------------------
__global__ void prep_A_kernel(const float* __restrict__ Ap,
                              float* __restrict__ A2, int n) {
    int i = blockIdx.x * 256 + threadIdx.x;
    if (i < n) {
        float a  = Ap[i];
        float sp = (a > 20.f) ? a : log1pf(__expf(a));
        sp       = fminf(fmaxf(sp, 0.1f), 10.f);
        A2[i]    = -sp;
    }
}

// ---------------------------------------------------------------------------
// f32 -> f16 elementwise (n4 = count/4)
// ---------------------------------------------------------------------------
__global__ void cvt_f16_kernel(const float* __restrict__ s,
                               _Float16* __restrict__ d, int n4) {
    int i = blockIdx.x * 256 + threadIdx.x;
    if (i < n4) {
        float4 v = ((const float4*)s)[i];
        union { _Float16 h[4]; unsigned long long u; } p;
        p.h[0] = (_Float16)v.x; p.h[1] = (_Float16)v.y;
        p.h[2] = (_Float16)v.z; p.h[3] = (_Float16)v.w;
        *(unsigned long long*)(d + 4 * (size_t)i) = p.u;
    }
}

// ---------------------------------------------------------------------------
// W (KxN f32, row-major) -> Bt (NxK f16, row-major) transpose+convert.
// 32x32 tiles through LDS; both global accesses coalesced. K%32==N%32==0.
// ---------------------------------------------------------------------------
__global__ __launch_bounds__(256)
void transpose_cvt_kernel(const float* __restrict__ W,
                          _Float16* __restrict__ Bt, int K, int N) {
    __shared__ float t[32][33];
    const int n0 = blockIdx.x * 32, k0 = blockIdx.y * 32;
    const int tx = threadIdx.x & 31, ty = threadIdx.x >> 5;   // 32 x 8
    #pragma unroll
    for (int r = 0; r < 32; r += 8)
        t[ty + r][tx] = W[(size_t)(k0 + ty + r) * N + n0 + tx];
    __syncthreads();
    #pragma unroll
    for (int r = 0; r < 32; r += 8)
        Bt[(size_t)(n0 + ty + r) * K + k0 + tx] = (_Float16)t[tx][ty + r];
}

// ---------------------------------------------------------------------------
// WMMA GEMM: C(MxN) f32 = A(MxK) f16 * Bt(NxK) f16   (Bt is B transposed)
// Block tile 128x128, K-tile 32, 256 threads = 8 waves; wave = 32x64 = 8 wmma.
// Double-buffered LDS filled with GLOBAL_LOAD_ASYNC_TO_LDS_B128 (ASYNCcnt).
// Requires M%128==0, N%128==0, K%32==0 (true for all shapes here).
// ---------------------------------------------------------------------------
__global__ __launch_bounds__(256)
void gemm_wmma_f16t(const _Float16* __restrict__ A, const _Float16* __restrict__ Bt,
                    float* __restrict__ C, int M, int N, int K) {
    __shared__ _Float16 As[2][128 * 32];   // [m][k], k contiguous
    __shared__ _Float16 Bs[2][128 * 32];   // [n][k], k contiguous

    const int tid  = threadIdx.x;
    const int lane = tid & 31;          // wave32
    const int wave = tid >> 5;          // 0..7
    const int wr   = wave >> 1;         // 0..3 : 32-row stripe
    const int wc   = wave & 1;          // 0..1 : 64-col stripe
    const int hi   = (lane >> 4) & 1;
    const int l15  = lane & 15;
    const int bm   = blockIdx.y * 128;
    const int bn   = blockIdx.x * 128;

    // LDS byte offsets of the buffers (generic ptr low 32 bits = LDS offset)
    const unsigned asbase = (unsigned)(uintptr_t)&As[0][0];
    const unsigned bsbase = (unsigned)(uintptr_t)&Bs[0][0];

    v8f zero = {};
    v8f acc[2][4];
    #pragma unroll
    for (int a = 0; a < 2; ++a)
        #pragma unroll
        for (int b = 0; b < 4; ++b) acc[a][b] = zero;

    union Frag { v16h v; unsigned u[8]; };

    // issue one tile copy (A: 128x32 halves = 8KB, B same). Each thread moves
    // 2x16B of A and 2x16B of B -> 4 async instructions per wave per tile.
    auto copy_tile = [&](int buf, int k0) {
        #pragma unroll
        for (int r = 0; r < 2; ++r) {
            int q   = tid + 256 * r;        // 0..511 16-byte chunks
            int row = q >> 2;               // 0..127
            int seg = (q & 3) * 8;          // halves within row
            async_cp_b128(asbase + (unsigned)buf * 8192u + (unsigned)q * 16u,
                          A + (size_t)(bm + row) * K + k0 + seg);
            async_cp_b128(bsbase + (unsigned)buf * 8192u + (unsigned)q * 16u,
                          Bt + (size_t)(bn + row) * K + k0 + seg);
        }
    };

    copy_tile(0, 0);
    const int nk = K >> 5;
    for (int kt = 0; kt < nk; ++kt) {
        const int cur = kt & 1;
        const bool more = (kt + 1) < nk;
        if (more) copy_tile(1 - cur, (kt + 1) << 5);
        if (more) wait_async_le4();     // current tile's 4 copies retired
        else      wait_async_0();
        __syncthreads();

        const _Float16* Ab = &As[cur][0];
        const _Float16* Bb = &Bs[cur][0];
        Frag fa[2], fb[4];
        #pragma unroll
        for (int t = 0; t < 2; ++t) {
            int m = wr * 32 + t * 16 + l15;
            #pragma unroll
            for (int j = 0; j < 8; ++j) {   // ISA 7.12.2 16-bit A layout
                int kp = ((j < 4) ? (2 * j) : (16 + 2 * (j - 4))) + hi * 8;
                fa[t].u[j] = *(const unsigned*)&Ab[m * 32 + kp];
            }
        }
        #pragma unroll
        for (int t = 0; t < 4; ++t) {
            int n = wc * 64 + t * 16 + l15;
            #pragma unroll
            for (int j = 0; j < 8; ++j) {   // B: same K order as A
                int kp = ((j < 4) ? (2 * j) : (16 + 2 * (j - 4))) + hi * 8;
                fb[t].u[j] = *(const unsigned*)&Bb[n * 32 + kp];
            }
        }
        #pragma unroll
        for (int ti = 0; ti < 2; ++ti)
            #pragma unroll
            for (int tj = 0; tj < 4; ++tj)
                acc[ti][tj] = __builtin_amdgcn_wmma_f32_16x16x32_f16(
                    false, fa[ti].v, false, fb[tj].v,
                    (short)0, acc[ti][tj], false, false);
        __syncthreads();   // protects buffer `cur` (overwritten next iter)
    }

    // store D: lanes carry N; VGPR r -> M = r + hi*8
    #pragma unroll
    for (int ti = 0; ti < 2; ++ti) {
        #pragma unroll
        for (int tj = 0; tj < 4; ++tj) {
            int n = bn + wc * 64 + tj * 16 + l15;
            #pragma unroll
            for (int r = 0; r < 8; ++r) {
                int m = bm + wr * 32 + ti * 16 + r + hi * 8;
                C[(size_t)m * N + n] = acc[ti][tj][r];
            }
        }
    }
}

// ---------------------------------------------------------------------------
// Kernel: per (b,s) row — causal depthwise conv(K=4) + SiLU + LayerNorm(I)
//         + xp = xc@W_xproj (33 outs) + SSM gate + y*silu(z) -> U (f16)
// ---------------------------------------------------------------------------
__global__ __launch_bounds__(256)
void fuse_mid_kernel(const float* __restrict__ xz, const float* __restrict__ convw,
                     const float* __restrict__ gamma, const float* __restrict__ beta,
                     const float* __restrict__ Wx, const float* __restrict__ A2,
                     const float* __restrict__ Dv, _Float16* __restrict__ U) {
    const int row  = blockIdx.x;        // b*S + s
    const int sidx = row & (S_ - 1);
    const int tid  = threadIdx.x;
    const int lane = tid & 31;
    const int wid  = tid >> 5;

    __shared__ float redA[8], redB[8];
    __shared__ float red33[8][33];
    __shared__ float xpL[33];
    __shared__ float stats[2];

    float xc[8];
    #pragma unroll
    for (int c = 0; c < 8; ++c) {       // causal conv + SiLU
        int i = tid + 256 * c;
        float a = 0.f;
        #pragma unroll
        for (int t = 0; t < 4; ++t) {
            int sp  = sidx - 3 + t;
            float v = (sp >= 0) ? xz[(size_t)(row - 3 + t) * TWO_I + i] : 0.f;
            a += v * convw[i * 4 + t];
        }
        xc[c] = a / (1.f + __expf(-a));
    }
    float s1 = 0.f, s2 = 0.f;           // LN stats
    #pragma unroll
    for (int c = 0; c < 8; ++c) { s1 += xc[c]; s2 += xc[c] * xc[c]; }
    for (int o = 16; o > 0; o >>= 1) {
        s1 += __shfl_down(s1, o, 32);
        s2 += __shfl_down(s2, o, 32);
    }
    if (lane == 0) { redA[wid] = s1; redB[wid] = s2; }
    __syncthreads();
    if (tid == 0) {
        float t1 = 0.f, t2 = 0.f;
        for (int w = 0; w < 8; ++w) { t1 += redA[w]; t2 += redB[w]; }
        float mu  = t1 / (float)I_;
        float var = t2 / (float)I_ - mu * mu;
        stats[0] = mu;
        stats[1] = rsqrtf(var + 1e-5f);
    }
    __syncthreads();
    const float mu = stats[0], rstd = stats[1];
    #pragma unroll
    for (int c = 0; c < 8; ++c) {
        int i = tid + 256 * c;
        xc[c] = (xc[c] - mu) * rstd * gamma[i] + beta[i];
    }
    float xp[33];                       // xc @ W_xproj
    #pragma unroll
    for (int j = 0; j < 33; ++j) xp[j] = 0.f;
    #pragma unroll
    for (int c = 0; c < 8; ++c) {
        int i = tid + 256 * c;
        const float* wr2 = Wx + (size_t)i * 33;
        float xv = xc[c];
        #pragma unroll
        for (int j = 0; j < 33; ++j) xp[j] += xv * wr2[j];
    }
    #pragma unroll
    for (int j = 0; j < 33; ++j) {
        float v = xp[j];
        for (int o = 16; o > 0; o >>= 1) v += __shfl_down(v, o, 32);
        if (lane == 0) red33[wid][j] = v;
    }
    __syncthreads();
    if (tid < 33) {
        float v = 0.f;
        for (int w = 0; w < 8; ++w) v += red33[w][tid];
        xpL[tid] = v;
    }
    __syncthreads();
    float sp0   = xpL[0];
    float spl   = (sp0 > 20.f) ? sp0 : log1pf(__expf(sp0));
    float delta = fminf(fmaxf(spl, 1e-6f), 10.f);
    float wbc[NST];
    #pragma unroll
    for (int n = 0; n < NST; ++n) wbc[n] = xpL[1 + n] * xpL[1 + NST + n];
    #pragma unroll
    for (int c = 0; c < 8; ++c) {
        int i = tid + 256 * c;
        const float* ar = A2 + (size_t)i * NST;
        float s = 0.f;
        #pragma unroll
        for (int n = 0; n < NST; ++n) {
            float d = __expf(delta * ar[n]);
            d = fminf(fmaxf(d, 1e-6f), 1.f);
            s += wbc[n] * d;
        }
        float y = xc[c] * s + xc[c] * Dv[i];
        float z = xz[(size_t)row * TWO_I + I_ + i];
        U[(size_t)row * I_ + i] = (_Float16)(y * (z / (1.f + __expf(-z))));
    }
}

// ---------------------------------------------------------------------------
extern "C" void kernel_launch(void* const* d_in, const int* in_sizes, int n_in,
                              void* d_out, int out_size, void* d_ws, size_t ws_size,
                              hipStream_t stream) {
    const float* x      = (const float*)d_in[0];
    const float* W_in   = (const float*)d_in[1];
    const float* conv_w = (const float*)d_in[2];
    const float* ln_g   = (const float*)d_in[3];
    const float* ln_b   = (const float*)d_in[4];
    const float* W_x    = (const float*)d_in[5];
    const float* A_p    = (const float*)d_in[6];
    const float* Dv     = (const float*)d_in[7];
    const float* W_out  = (const float*)d_in[8];
    float* out = (float*)d_out;

    char* ws = (char*)d_ws;
    size_t off = 0;
    float*    xz  = (float*)(ws + off);    off += (size_t)ROWS * TWO_I * 4;  // 64MB
    _Float16* Uh  = (_Float16*)(ws + off); off += (size_t)ROWS * I_ * 2;     // 16MB
    _Float16* Ah  = (_Float16*)(ws + off); off += (size_t)ROWS * H_ * 2;     //  8MB
    _Float16* B1t = (_Float16*)(ws + off); off += (size_t)TWO_I * H_ * 2;    //  8MB
    _Float16* B2t = (_Float16*)(ws + off); off += (size_t)H_ * I_ * 2;       //  4MB
    float*    A2  = (float*)(ws + off);

    // prep: gate table, f16 A operand, f16 transposed B operands
    prep_A_kernel<<<(I_ * NST + 255) / 256, 256, 0, stream>>>(A_p, A2, I_ * NST);
    cvt_f16_kernel<<<(ROWS * H_ / 4 + 255) / 256, 256, 0, stream>>>(x, Ah, ROWS * H_ / 4);
    transpose_cvt_kernel<<<dim3(TWO_I / 32, H_ / 32), 256, 0, stream>>>(W_in, B1t, H_, TWO_I);
    transpose_cvt_kernel<<<dim3(H_ / 32, I_ / 32), 256, 0, stream>>>(W_out, B2t, I_, H_);

    // xz = x @ W_in   (M=4096, N=4096, K=1024)
    gemm_wmma_f16t<<<dim3(TWO_I / 128, ROWS / 128), 256, 0, stream>>>(
        Ah, B1t, xz, ROWS, TWO_I, H_);

    fuse_mid_kernel<<<ROWS, 256, 0, stream>>>(xz, conv_w, ln_g, ln_b,
                                              W_x, A2, Dv, Uh);

    // out = U @ W_out (M=4096, N=1024, K=2048)
    gemm_wmma_f16t<<<dim3(H_ / 128, ROWS / 128), 256, 0, stream>>>(
        Uh, B2t, out, ROWS, H_, I_);
}
